// FilterBankLayer_18777597018862
// MI455X (gfx1250) — compile-verified, matrix operand order checked by reading, not run on hardware
//
#include <hip/hip_runtime.h>

// FilterBank depthwise conv as im2col GEMM on CDNA5 WMMA (bf16 hi/lo split, fp32 accum).
//
// Shapes: x (64,64,1000) f32, kernels (9,125) f32, out (64,9,64,1000) f32.
// Tile: 16 timesteps x 16 bands (9 valid), K padded 125 -> 128 (4 WMMA K-steps of 32).

typedef __attribute__((ext_vector_type(16))) __bf16 v16bf;
typedef __attribute__((ext_vector_type(8)))  float  v8f;

#define T_LEN   1000
#define NCH     64
#define NBATCH  64
#define NB      9
#define KLEN    125
#define PADL    62          // KLEN/2
#define XLDS    1136        // t0max(992)+15 + kmax(127) - 62 = 1072; +halo 62 = 1134 -> 1136
#define NWAVE   8
#define NTILE   63          // ceil(1000/16)

__global__ __launch_bounds__(256)
void fbank_wmma_kernel(const float* __restrict__ x,
                       const float* __restrict__ kern,
                       float* __restrict__ out)
{
    __shared__ __bf16 xh[XLDS];            // bf16 high part of padded signal row
    __shared__ __bf16 xl[XLDS];            // bf16 residual (x - hi)
    __shared__ float  stg[NWAVE][16][17];  // per-wave transpose staging, padded stride

    const int tid  = threadIdx.x;
    const int lane = tid & 31;
    const int wave = tid >> 5;
    const int row  = blockIdx.x;           // b*64 + c
    const int b    = row >> 6;
    const int c    = row & 63;

    // ---- Stage one (b,c) signal row into LDS with zero halo, split hi/lo ----
    for (int i = tid; i < XLDS; i += 256) {
        const int t = i - PADL;
        float v = (t >= 0 && t < T_LEN) ? x[row * T_LEN + t] : 0.0f;
        __bf16 h = (__bf16)v;
        xh[i] = h;
        xl[i] = (__bf16)(v - (float)h);
    }

    // ---- Build B fragments (kernels^T, 128x16) in registers, hi/lo, 4 K-steps ----
    // 16-bit operand lane layout: lanes 0-15 / 16-31 split; element e -> K offset
    // koff(e) = e + (e>=8 ? 8 : 0) + (lane>=16 ? 8 : 0)   (per ISA 7.12.2 layout)
    const int nN  = lane & 15;             // N (band) for B; M (time) for A
    const int lhi = lane >> 4;
    v16bf bh[4], bl[4];
    #pragma unroll
    for (int s = 0; s < 4; ++s) {
        #pragma unroll
        for (int e = 0; e < 16; ++e) {
            const int k = s * 32 + e + ((e >= 8) ? 8 : 0) + (lhi ? 8 : 0);
            float v = (nN < NB && k < KLEN) ? kern[nN * KLEN + k] : 0.0f;
            __bf16 h = (__bf16)v;
            bh[s][e] = h;
            bl[s][e] = (__bf16)(v - (float)h);
        }
    }

    __syncthreads();

    // ---- Tiles of 16 output timesteps; wave-uniform loop so EXEC==all-1s at WMMA ----
    for (int tile = wave; tile < NTILE; tile += NWAVE) {
        const int t0 = tile * 16;
        v8f acc = {};

        #pragma unroll
        for (int s = 0; s < 4; ++s) {
            // A fragment (16x32 bf16): A[m, k] = x[t0 + m + k - 62] = xlds[t0 + m + k]
            v16bf ah, al;
            const int base = t0 + nN + s * 32 + (lhi ? 8 : 0);
            #pragma unroll
            for (int e = 0; e < 16; ++e) {
                const int idx = base + e + ((e >= 8) ? 8 : 0);
                ah[e] = xh[idx];
                al[e] = xl[idx];
            }
            // (Ah+Al)(Bh+Bl) ~= AhBh + AhBl + AlBh, fp32 accumulate
            acc = __builtin_amdgcn_wmma_f32_16x16x32_bf16(false, ah, false, bh[s],
                                                          (short)0, acc, false, false);
            acc = __builtin_amdgcn_wmma_f32_16x16x32_bf16(false, ah, false, bl[s],
                                                          (short)0, acc, false, false);
            acc = __builtin_amdgcn_wmma_f32_16x16x32_bf16(false, al, false, bh[s],
                                                          (short)0, acc, false, false);
        }

        // ---- Transpose through LDS: C/D layout lane=N, VGPR r -> M = r + 8*lhi ----
        #pragma unroll
        for (int r = 0; r < 8; ++r)
            stg[wave][r + (lhi ? 8 : 0)][nN] = acc[r];

        // ---- Coalesced store: 9 bands x 16 contiguous timesteps ----
        #pragma unroll
        for (int it = 0; it < 5; ++it) {            // ceil(9*16/32)
            const int idx = it * 32 + lane;
            const int nb  = idx >> 4;
            const int m   = idx & 15;
            const int t   = t0 + m;
            if (nb < NB && t < T_LEN)
                out[((b * NB + nb) * NCH + c) * T_LEN + t] = stg[wave][m][nb];
        }
    }
}

extern "C" void kernel_launch(void* const* d_in, const int* in_sizes, int n_in,
                              void* d_out, int out_size, void* d_ws, size_t ws_size,
                              hipStream_t stream)
{
    const float* x    = (const float*)d_in[0];   // (64,64,1000) f32
    const float* kern = (const float*)d_in[1];   // (9,125) f32
    float*       out  = (float*)d_out;           // (64,9,64,1000) f32

    fbank_wmma_kernel<<<dim3(NBATCH * NCH), dim3(256), 0, stream>>>(x, kern, out);
}
